// SS2D_intra_1597727834516
// MI455X (gfx1250) — matrix operand order repaired and checked by reading, not compiled, not run on hardware
//
#include <hip/hip_runtime.h>
#include <hip/hip_bf16.h>

typedef __attribute__((ext_vector_type(16))) _Float16 v16h;
typedef __attribute__((ext_vector_type(8)))  float    v8f;
typedef __attribute__((address_space(3))) float lds_float;

#define B_  32
#define N_  128
#define D_  512
#define DI_ 1024
#define DS_ 16
#define DR_ 32
#define H_  16
#define W_  8
#define BN_ (B_ * N_)   /* 4096 */
#define P_  (3 * N_)    /* 384  */

// ---- f32 -> f16 tile loaders (b128 loads, convert in-register) ----
__device__ __forceinline__ void load8h(const float* p, v16h& a, int off) {
  const float4* q = reinterpret_cast<const float4*>(p);
  float4 x0 = q[0], x1 = q[1];
  a[off + 0] = (_Float16)x0.x; a[off + 1] = (_Float16)x0.y;
  a[off + 2] = (_Float16)x0.z; a[off + 3] = (_Float16)x0.w;
  a[off + 4] = (_Float16)x1.x; a[off + 5] = (_Float16)x1.y;
  a[off + 6] = (_Float16)x1.z; a[off + 7] = (_Float16)x1.w;
}

// One K=32 step of D = A(16xK) * W^T(Kx16) + C via v_wmma_f32_16x16x32_f16.
__device__ __forceinline__ v8f gemm_kstep(const float* __restrict__ Arow,
                                          const float* __restrict__ Wrow,
                                          int k0, int half, v8f acc) {
  v16h a, b;
  // A 16x32 f16 layout: lanes<16 slots = K{0..7,16..23}, lanes>=16 = K{8..15,24..31}
  load8h(Arow + k0 + 8 * half,      a, 0);
  load8h(Arow + k0 + 16 + 8 * half, a, 8);
  // B 32x16 f16 layout: lanes<16 slots = K{0..7,16..23}, lanes>=16 = K{8..15,24..31}
  load8h(Wrow + k0 + 8 * half,      b, 0);
  load8h(Wrow + k0 + 16 + 8 * half, b, 8);
  return __builtin_amdgcn_wmma_f32_16x16x32_f16(false, a, false, b,
                                                (short)0, acc, false, false);
}

// Async global->LDS b128 copy (ASYNCcnt-tracked, CDNA5 §15.18 op 98).
__device__ __forceinline__ void async_copy_b128(uint32_t lds_off, const float* g) {
  asm volatile("global_load_async_to_lds_b128 %0, %1, off"
               :: "v"(lds_off), "v"(g) : "memory");
}
__device__ __forceinline__ void wait_asynccnt0() {
  asm volatile("s_wait_asynccnt 0x0" ::: "memory");
}

// ---------------- 1) in_proj: xz[m] = ins[m] @ in_proj_w[m]^T ----------------
__global__ void k_inproj(const float* __restrict__ r, const float* __restrict__ n,
                         const float* __restrict__ t, const float* __restrict__ w,
                         float* __restrict__ xz) {
  const int lane = threadIdx.x & 31, wave = threadIdx.x >> 5;
  const int half = lane >> 4, l15 = lane & 15;
  const int n0 = (blockIdx.x * 8 + wave) * 16;
  const int m0 = blockIdx.y * 16;
  const int m  = blockIdx.z;
  const float* X = (m == 0) ? r : (m == 1) ? n : t;
  const float* Arow = X + (size_t)(m0 + l15) * D_;
  const float* Wrow = w + (size_t)m * (2 * DI_) * D_ + (size_t)(n0 + l15) * D_;
  v8f acc = {};
#pragma unroll 4
  for (int k0 = 0; k0 < D_; k0 += 32) {
    __builtin_prefetch(Arow + k0 + 128, 0, 0);
    __builtin_prefetch(Wrow + k0 + 128, 0, 0);
    acc = gemm_kstep(Arow, Wrow, k0, half, acc);
  }
  float* out = xz + (size_t)(m * BN_) * (2 * DI_);
#pragma unroll
  for (int v = 0; v < 8; ++v)
    out[(size_t)(m0 + v + 8 * half) * (2 * DI_) + n0 + l15] = acc[v];
}

// ------------- 2) depthwise 3x3 conv + BN + SiLU (reads x-half of xz) -------
__global__ void k_conv(const float* __restrict__ xz, const float* __restrict__ cw,
                       const float* __restrict__ cb, const float* __restrict__ gm,
                       const float* __restrict__ bt, const float* __restrict__ mn,
                       const float* __restrict__ vr, float* __restrict__ xconv) {
  int g = blockIdx.x * blockDim.x + threadIdx.x;
  int c = g & (DI_ - 1);
  int rest = g >> 10;
  int nn = rest & (N_ - 1);
  int b = (rest >> 7) & (B_ - 1);
  int m = rest >> 12;
  int h = nn >> 3, ww = nn & 7;
  const float* wp = cw + ((size_t)m * DI_ + c) * 9;
  float acc = 0.f;
#pragma unroll
  for (int dh = -1; dh <= 1; ++dh) {
    int hh = h + dh;
    if (hh < 0 || hh >= H_) continue;
#pragma unroll
    for (int dw = -1; dw <= 1; ++dw) {
      int w2 = ww + dw;
      if (w2 < 0 || w2 >= W_) continue;
      float xv = xz[(size_t)(m * BN_ + b * N_ + hh * W_ + w2) * (2 * DI_) + c];
      acc += xv * wp[(dh + 1) * 3 + (dw + 1)];
    }
  }
  int mc = m * DI_ + c;
  acc += cb[mc];
  acc = (acc - mn[mc]) * (gm[mc] * rsqrtf(vr[mc] + 1e-5f)) + bt[mc];
  acc = acc / (1.f + expf(-acc));  // SiLU
  xconv[(size_t)(m * BN_ + b * N_ + nn) * DI_ + c] = acc;
}

// -- 3) xproj: xd[m6] = xs6[m6] @ xproj_w[m6]^T  (flip folded into A rows) ---
__global__ void k_xproj(const float* __restrict__ xconv,
                        const float* __restrict__ xw, float* __restrict__ xd) {
  const int lane = threadIdx.x & 31, wave = threadIdx.x >> 5;  // block=128
  const int half = lane >> 4, l15 = lane & 15;
  const int n0 = wave * 16;
  const int mtile = blockIdx.x;
  const int m6 = blockIdx.y;
  const int m = (m6 < 3) ? m6 : m6 - 3;
  const bool flip = m6 >= 3;
  int ri = mtile * 16 + l15;
  int b = ri >> 7, nl = ri & (N_ - 1);
  int ns = flip ? (N_ - 1 - nl) : nl;
  const float* Arow = xconv + (size_t)(m * BN_ + b * N_ + ns) * DI_;
  const float* Wrow = xw + (size_t)m6 * 64 * DI_ + (size_t)(n0 + l15) * DI_;
  v8f acc = {};
#pragma unroll 4
  for (int k0 = 0; k0 < DI_; k0 += 32) {
    __builtin_prefetch(Arow + k0 + 128, 0, 0);
    acc = gemm_kstep(Arow, Wrow, k0, half, acc);
  }
  float* out = xd + (size_t)(m6 * BN_) * 64;
#pragma unroll
  for (int v = 0; v < 8; ++v)
    out[(size_t)(mtile * 16 + v + 8 * half) * 64 + n0 + l15] = acc[v];
}

// ---- 4) delta = softplus(dt @ dtw^T + dtb); single K=32 WMMA per tile ------
__global__ void k_dt(const float* __restrict__ xd, const float* __restrict__ dtw,
                     const float* __restrict__ dtb, float* __restrict__ delta) {
  const int lane = threadIdx.x & 31, wave = threadIdx.x >> 5;
  const int half = lane >> 4, l15 = lane & 15;
  const int n0 = (blockIdx.x * 8 + wave) * 16;
  const int m0 = blockIdx.y * 16;
  const int m6 = blockIdx.z;
  const float* Arow = xd + (size_t)(m6 * BN_ + m0 + l15) * 64;
  const float* Wrow = dtw + (size_t)m6 * DI_ * DR_ + (size_t)(n0 + l15) * DR_;
  v8f acc = {};
  acc = gemm_kstep(Arow, Wrow, 0, half, acc);
#pragma unroll
  for (int v = 0; v < 8; ++v) {
    int row = m0 + v + 8 * half, j = n0 + l15;
    float x = acc[v] + dtb[m6 * DI_ + j];
    float sp = (x > 20.f) ? x : log1pf(expf(x));
    delta[(size_t)(m6 * BN_ + row) * DI_ + j] = sp;
  }
}

// ------- 5) selective scan: thread per (m6,b,d), h[16] in VGPRs -------------
__global__ void k_scan(const float* __restrict__ xd, const float* __restrict__ xconv,
                       const float* __restrict__ A_log, const float* __restrict__ Dp,
                       float* __restrict__ dy) {
  int g = blockIdx.x * blockDim.x + threadIdx.x;
  int d = g & (DI_ - 1);
  int rest = g >> 10;
  int b = rest & (B_ - 1);
  int m6 = rest >> 5;
  int m = (m6 < 3) ? m6 : m6 - 3;
  bool flip = m6 >= 3;
  float A[DS_], h[DS_];
#pragma unroll
  for (int s = 0; s < DS_; ++s) {
    A[s] = -expf(A_log[((size_t)m6 * DI_ + d) * DS_ + s]);
    h[s] = 0.f;
  }
  float Dv = Dp[m6 * DI_ + d];
  for (int nl = 0; nl < N_; ++nl) {
    size_t bn = (size_t)(m6 * BN_ + b * N_ + nl);
    float dv = dy[bn * DI_ + d];
    int ns = flip ? (N_ - 1 - nl) : nl;
    float u = xconv[(size_t)(m * BN_ + b * N_ + ns) * DI_ + d];
    const float4* q = reinterpret_cast<const float4*>(xd + bn * 64 + 32);
    float4 b0 = q[0], b1 = q[1], b2 = q[2], b3 = q[3];
    float4 c0 = q[4], c1 = q[5], c2 = q[6], c3 = q[7];
    float Bt[DS_] = {b0.x, b0.y, b0.z, b0.w, b1.x, b1.y, b1.z, b1.w,
                     b2.x, b2.y, b2.z, b2.w, b3.x, b3.y, b3.z, b3.w};
    float Ct[DS_] = {c0.x, c0.y, c0.z, c0.w, c1.x, c1.y, c1.z, c1.w,
                     c2.x, c2.y, c2.z, c2.w, c3.x, c3.y, c3.z, c3.w};
    float du = dv * u, y = 0.f;
#pragma unroll
    for (int s = 0; s < DS_; ++s) {
      h[s] = expf(dv * A[s]) * h[s] + du * Bt[s];
      y += h[s] * Ct[s];
    }
    dy[bn * DI_ + d] = y + u * Dv;
  }
}

// ----------------- 6) LayerNorm over DI per (m6,b,n) row, in place ----------
__global__ void k_lnorm(float* __restrict__ y, const float* __restrict__ g,
                        const float* __restrict__ b) {
  __shared__ float s1[256], s2[256];
  int row = blockIdx.x;
  int m6 = row >> 12;
  float* p = y + (size_t)row * DI_;
  int tid = threadIdx.x;
  float sum = 0.f, sq = 0.f;
  for (int i = tid; i < DI_; i += 256) { float v = p[i]; sum += v; sq += v * v; }
  s1[tid] = sum; s2[tid] = sq;
  __syncthreads();
  for (int off = 128; off > 0; off >>= 1) {
    if (tid < off) { s1[tid] += s1[tid + off]; s2[tid] += s2[tid + off]; }
    __syncthreads();
  }
  float mu = s1[0] * (1.f / DI_);
  float var = s2[0] * (1.f / DI_) - mu * mu;
  float rstd = rsqrtf(var + 1e-5f);
  for (int i = tid; i < DI_; i += 256)
    p[i] = (p[i] - mu) * rstd * g[m6 * DI_ + i] + b[m6 * DI_ + i];
}

// --------- 7) fwd+bwd combine, SiLU(z) gate -> ycat (B, 384, DI) ------------
__global__ void k_gate(const float* __restrict__ ybuf, const float* __restrict__ xz,
                       float* __restrict__ ycat) {
  int g = blockIdx.x * blockDim.x + threadIdx.x;
  int d = g & (DI_ - 1);
  int rest = g >> 10;
  int p = rest % P_, b = rest / P_;
  int seg = p >> 7, nl = p & (N_ - 1);
  float yf = ybuf[(size_t)(seg * BN_ + b * N_ + nl) * DI_ + d];
  float yb = ybuf[(size_t)((seg + 3) * BN_ + b * N_ + (N_ - 1 - nl)) * DI_ + d];
  float y = yf + yb;
  float zc = xz[(size_t)(seg * BN_ + b * N_ + nl) * (2 * DI_) + DI_ + d];
  y *= zc / (1.f + expf(-zc));
  ycat[(size_t)(b * P_ + p) * DI_ + d] = y;
}

// ------- 8) out_proj GEMM (12288x1024)*(1024x512)^T, residual fused ---------
// A-tile (16x32 per K-step) staged once per block into LDS with async b128
// copies, ping-pong double buffered; 8 waves share it (8x less A traffic).
#define SROW 36                      /* padded row: 36*L mod 64 distinct */
#define KSTEPS (DI_ / 32)
__global__ void k_out(const float* __restrict__ ycat, const float* __restrict__ ow,
                      const float* __restrict__ r, const float* __restrict__ n,
                      const float* __restrict__ t, float* __restrict__ out) {
  __shared__ __align__(16) float sA[2][16 * SROW];
  const int tid = threadIdx.x;
  const int lane = tid & 31, wave = tid >> 5;
  const int half = lane >> 4, l15 = lane & 15;
  const int nb = blockIdx.x * 128;          // grid.x = 4 -> 512 cols
  const int n0 = nb + wave * 16;            // 8 waves x 16 cols
  const int m0 = blockIdx.y * 16;           // grid.y = 768 -> 12288 rows
  const float* Wrow = ow + (size_t)(n0 + l15) * DI_;
  // staging role: threads 0..127 move 16 rows x 32 floats (128 x b128)
  const int srow = tid >> 3, sq = tid & 7;
  const float* gA = ycat + (size_t)(m0 + srow) * DI_ + sq * 4;
  uint32_t lbase = (uint32_t)(uintptr_t)(lds_float*)&sA[0][0];
  uint32_t ldst = lbase + (uint32_t)(srow * SROW + sq * 4) * 4u;
  const uint32_t bufstride = 16 * SROW * 4;

  if (tid < 128) async_copy_b128(ldst, gA);            // stage chunk 0
  v8f acc = {};
  for (int step = 0; step < KSTEPS; ++step) {
    wait_asynccnt0();           // issuing waves: chunk `step` landed in LDS
    __syncthreads();            // publish to all waves; prior buffer free
    int cur = step & 1;
    if (step + 1 < KSTEPS && tid < 128)                // prefetch chunk step+1
      async_copy_b128(ldst + (uint32_t)((step + 1) & 1) * bufstride,
                      gA + (step + 1) * 32);
    const float* Ar = &sA[cur][l15 * SROW];
    v16h a, b;
    load8h(Ar + 8 * half, a, 0);                       // ds_load_b128 x2
    load8h(Ar + 16 + 8 * half, a, 8);
    int k0 = step * 32;
    load8h(Wrow + k0 + 8 * half, b, 0);
    load8h(Wrow + k0 + 16 + 8 * half, b, 8);
    acc = __builtin_amdgcn_wmma_f32_16x16x32_f16(false, a, false, b,
                                                 (short)0, acc, false, false);
  }
#pragma unroll
  for (int v = 0; v < 8; ++v) {
    int row = m0 + v + 8 * half;
    int b2 = row / P_, p = row % P_;
    int seg = p >> 7, nl = p & (N_ - 1);
    const float* res = (seg == 0) ? r : (seg == 1) ? n : t;
    size_t idx = (size_t)(b2 * N_ + nl) * D_ + n0 + l15;
    out[(size_t)seg * (B_ * N_ * D_) + idx] = res[idx] + acc[v];
  }
}

extern "C" void kernel_launch(void* const* d_in, const int* in_sizes, int n_in,
                              void* d_out, int out_size, void* d_ws, size_t ws_size,
                              hipStream_t stream) {
  (void)in_sizes; (void)n_in; (void)out_size; (void)ws_size;
  const float* r        = (const float*)d_in[0];
  const float* nn       = (const float*)d_in[1];
  const float* tt       = (const float*)d_in[2];
  const float* in_projw = (const float*)d_in[3];
  const float* conv_w   = (const float*)d_in[4];
  const float* conv_b   = (const float*)d_in[5];
  const float* bn_gamma = (const float*)d_in[6];
  const float* bn_beta  = (const float*)d_in[7];
  const float* bn_mean  = (const float*)d_in[8];
  const float* bn_var   = (const float*)d_in[9];
  const float* xproj_w  = (const float*)d_in[10];
  const float* dt_w     = (const float*)d_in[11];
  const float* dt_b     = (const float*)d_in[12];
  const float* A_log    = (const float*)d_in[13];
  const float* ssm_D    = (const float*)d_in[14];
  const float* ln_g     = (const float*)d_in[15];
  const float* ln_b     = (const float*)d_in[16];
  const float* out_w    = (const float*)d_in[17];
  float* out = (float*)d_out;

  float* ws    = (float*)d_ws;
  float* xz    = ws;                              // 3*4096*2048
  float* xconv = xz + (size_t)3 * BN_ * 2 * DI_;  // 3*4096*1024
  float* xd    = xconv + (size_t)3 * BN_ * DI_;   // 6*4096*64
  float* delta = xd + (size_t)6 * BN_ * 64;       // 6*4096*1024 (delta -> y in place)
  float* ycat  = delta + (size_t)6 * BN_ * DI_;   // 32*384*1024

  k_inproj<<<dim3(16, 256, 3), 256, 0, stream>>>(r, nn, tt, in_projw, xz);
  k_conv<<<(3 * BN_ * DI_) / 256, 256, 0, stream>>>(xz, conv_w, conv_b, bn_gamma,
                                                    bn_beta, bn_mean, bn_var, xconv);
  k_xproj<<<dim3(256, 6), 128, 0, stream>>>(xconv, xproj_w, xd);
  k_dt<<<dim3(8, 256, 6), 256, 0, stream>>>(xd, dt_w, dt_b, delta);
  k_scan<<<(6 * B_ * DI_) / 256, 256, 0, stream>>>(xd, xconv, A_log, ssm_D, delta);
  k_lnorm<<<6 * BN_, 256, 0, stream>>>(delta, ln_g, ln_b);
  k_gate<<<(B_ * P_ * DI_) / 256, 256, 0, stream>>>(delta, xz, ycat);
  k_out<<<dim3(4, 768), 256, 0, stream>>>(ycat, out_w, r, nn, tt, out);
}
